// CosFace_18408229830797
// MI455X (gfx1250) — matrix operand order repaired and checked by reading, not compiled, not run on hardware
//
#include <hip/hip_runtime.h>
#include <math.h>
#include <stdint.h>

typedef __attribute__((ext_vector_type(16))) __bf16 v16bf;
typedef __attribute__((ext_vector_type(8)))  float  v8f;

// Problem constants (match reference)
constexpr int   Bn  = 1024;     // batch
constexpr int   Dn  = 512;      // feature dim
constexpr int   Cn  = 100000;   // classes
constexpr float Sf  = 64.0f;    // scale
constexpr float Mf  = 0.35f;    // margin
constexpr int   NKC = Dn / 32;  // 16 k-chunks of 32 per WMMA

// GEMM tiling
constexpr int ROWS_PB = 64;     // rows per block  -> 4 M-tiles per wave
constexpr int CLS_PB  = 128;    // classes per block -> 8 waves x 16 classes
constexpr int NCHUNK  = (Cn + CLS_PB - 1) / CLS_PB;  // 782

// Raw workgroup split-barrier (no memory-fence -> no s_wait_loadcnt 0 drain).
// Safe here because cross-wave LDS ordering is enforced explicitly by
// s_wait_asynccnt (producer side) and by the ds_load->WMMA dscnt waits the
// backend inserts (consumer side) before the next signal.
#define WG_BARRIER() \
  asm volatile("s_barrier_signal -1\n\ts_barrier_wait -1" ::: "memory")

// ---------------------------------------------------------------------------
// Kernel 1: L2-normalize 16 rows and emit bf16 packed in WMMA register order.
// aLayout=1 -> A-operand layout (16-bit A 16x32, ISA 7.12.2):
//              lane = m + 16*((kin>>3)&1), elem e = (kin&7) | ((kin&16)>>1)
// aLayout=0 -> B-operand layout (K x 16 column-per-lane):
//              lane = m + 16*((kin>>3)&1), elem e = (kin&7) | ((kin&16)>>1)
// Packed element index: (((tile*NKC + kc)*32 + lane)*16 + e)  (32B/lane loads)
// ---------------------------------------------------------------------------
__global__ __launch_bounds__(256)
void pack_l2norm_bf16(const float* __restrict__ src,
                      unsigned short* __restrict__ dst, int aLayout) {
  __shared__ float xs[16][512];   // 32 KB staging
  __shared__ float ssq[16];
  __shared__ float invn[16];
  const int tile = blockIdx.x, tid = threadIdx.x;
  if (tid < 16) ssq[tid] = 0.0f;
  __syncthreads();
  for (int i = tid; i < 16 * 512; i += 256) {
    const int m = i >> 9, k = i & 511;
    const float v = src[((size_t)tile * 16 + m) * 512 + k];
    xs[m][k] = v;
    atomicAdd(&ssq[m], v * v);      // ds_add_f32
  }
  __syncthreads();
  if (tid < 16) invn[tid] = 1.0f / fmaxf(sqrtf(ssq[tid]), 1e-12f);
  __syncthreads();
  for (int i = tid; i < 16 * 512; i += 256) {
    const int m = i >> 9, k = i & 511;
    const float v = xs[m][k] * invn[m];
    unsigned u = __float_as_uint(v);            // f32 -> bf16 RNE
    u += 0x7fffu + ((u >> 16) & 1u);
    const unsigned short h = (unsigned short)(u >> 16);
    const int kc = k >> 5, kin = k & 31;
    int hf, e;
    if (aLayout) { hf = (kin >> 3) & 1; e = (kin & 7) | ((kin & 16) >> 1); }
    else         { hf = (kin >> 3) & 1; e = (kin & 7) | ((kin & 16) >> 1); }
    const int laneo = m + 16 * hf;
    dst[(((size_t)tile * NKC + kc) * 32 + laneo) * 16 + e] = h;
  }
}

// ---------------------------------------------------------------------------
// Kernel 2: fused cosine GEMM (bf16 WMMA, f32 acc) + margin + per-chunk
// online-softmax statistics. Block = 64 rows x 128 classes, 256 threads.
//
// A operand (shared by all 8 waves) is staged through LDS with the CDNA5
// async copy path: one global_load_async_to_lds_b128 per wave per K-step
// (256 threads x 16B = the block's 4KB A chunk), double-buffered, with
// s_wait_asynccnt + raw split barriers. B operand stays in a register double
// buffer fed by global_load_b128 (+ prefetch); weight stream is L2-resident
// (102 MB bf16 inside the 192 MB L2).
// ---------------------------------------------------------------------------
__global__ __launch_bounds__(256)
void cosface_gemm(const unsigned short* __restrict__ pA,
                  const unsigned short* __restrict__ pW,
                  const int* __restrict__ label,
                  float* __restrict__ chunkMax,
                  float* __restrict__ chunkSum,
                  float* __restrict__ tgt) {
  __shared__ float lg[ROWS_PB][CLS_PB + 4];              // ~33.8 KB
  __shared__ int   lbl[ROWS_PB];
  __shared__ __align__(32) unsigned short aLds[2][4 * 32 * 16];  // 2 x 4KB

  const int tid  = threadIdx.x;
  const int wave = tid >> 5;
  const int lane = tid & 31;
  const int half = lane >> 4;
  const int nIdx = lane & 15;

  const int rowBase = blockIdx.x * ROWS_PB;
  const int cBase   = blockIdx.y * CLS_PB;

  if (tid < ROWS_PB) lbl[tid] = label[rowBase + tid];

  const int cTile   = (cBase >> 4) + wave;    // this wave's 16-class tile
  const int maxTile = Cn / 16 - 1;
  const int cTileC  = cTile < maxTile ? cTile : maxTile;  // clamp tail loads
  const int myClass = cTile * 16 + nIdx;
  const bool cvalid = myClass < Cn;

  const int aTile0  = rowBase >> 4;

  v8f acc[4];
  #pragma unroll
  for (int t = 0; t < 4; ++t) acc[t] = (v8f){0,0,0,0,0,0,0,0};

  const unsigned short* wPtr = pW + (((size_t)cTileC * NKC) * 32 + lane) * 16;

  // ---- async A staging setup: thread tid copies 16B of the 4KB chunk ----
  const int copyTile = tid >> 6;              // which of the 4 A tiles
  const int copyOff  = (tid & 63) * 16;       // byte offset within 1KB tile
  const char* aBytes = reinterpret_cast<const char*>(pA);
  // global byte address of (tile copyTile, kc=0) portion for this thread:
  const uint64_t gA0 = (uint64_t)(uintptr_t)(aBytes +
      ((size_t)(aTile0 + copyTile) * NKC) * 1024 + copyOff);
  const unsigned ldsDst0 =
      (unsigned)(uintptr_t)(&aLds[0][0]) + (unsigned)(tid * 16);
  const unsigned ldsDst1 =
      (unsigned)(uintptr_t)(&aLds[1][0]) + (unsigned)(tid * 16);

  // Prologue: start async fill of buffer 0 with kc=0, preload B for kc=0.
  {
    const uint64_t g = gA0;
    asm volatile("global_load_async_to_lds_b128 %0, %1, off"
                 :: "v"(ldsDst0), "v"(g) : "memory");
  }
  v16bf bmCur = *reinterpret_cast<const v16bf*>(wPtr);
  v16bf bmNxt;

  #pragma unroll
  for (int kc = 0; kc < NKC; ++kc) {
    WG_BARRIER();      // (a) all waves done reading the buffer we now refill
    if (kc + 1 < NKC) {
      const uint64_t g = gA0 + (uint64_t)(kc + 1) * 1024;
      const unsigned d = ((kc + 1) & 1) ? ldsDst1 : ldsDst0;
      asm volatile("global_load_async_to_lds_b128 %0, %1, off"
                   :: "v"(d), "v"(g) : "memory");
      bmNxt = *reinterpret_cast<const v16bf*>(wPtr + (size_t)(kc + 1) * 512);
    }
    if (kc + 2 < NKC)                 // keep the B stream warm in near caches
      __builtin_prefetch(wPtr + (size_t)(kc + 2) * 512, 0, 3);

    // Wait for this wave's kc-copy, then publish across the block.
    if (kc + 1 < NKC)
      asm volatile("s_wait_asynccnt 0x1" ::: "memory");
    else
      asm volatile("s_wait_asynccnt 0x0" ::: "memory");
    WG_BARRIER();      // (b) whole 4KB A chunk for kc is in LDS

    const unsigned short* aS = aLds[kc & 1];
    #pragma unroll
    for (int t = 0; t < 4; ++t) {
      const v16bf am =
          *reinterpret_cast<const v16bf*>(aS + (size_t)(t * 32 + lane) * 16);
      acc[t] = __builtin_amdgcn_wmma_f32_16x16x32_bf16(
          false, am, false, bmCur, (short)0, acc[t], false, false);
    }
    if (kc + 1 < NKC) bmCur = bmNxt;
  }

  __syncthreads();     // fenced: epilogue reuses LDS across waves

  // Epilogue: scale to logits, apply margin at label, stash tile in LDS.
  const int col = wave * 16 + nIdx;
  #pragma unroll
  for (int t = 0; t < 4; ++t) {
    #pragma unroll
    for (int r = 0; r < 8; ++r) {
      const int mrow = t * 16 + r + half * 8;  // C/D layout: VGPR r, half
      float val = cvalid ? Sf * acc[t][r] : -INFINITY;
      if (cvalid && lbl[mrow] == myClass) {
        val -= Sf * Mf;
        tgt[rowBase + mrow] = val;             // exactly one writer per row
      }
      lg[mrow][col] = val;
    }
  }
  __syncthreads();

  // Per-row chunk (max, sum-exp): 4 threads per row, shuffle-combine.
  const int rrow    = tid >> 2;
  const int colBase = (tid & 3) * 32;
  float mx = -INFINITY;
  #pragma unroll 8
  for (int j = 0; j < 32; ++j) mx = fmaxf(mx, lg[rrow][colBase + j]);
  mx = fmaxf(mx, __shfl_xor(mx, 1, 32));
  mx = fmaxf(mx, __shfl_xor(mx, 2, 32));
  float se = 0.0f;
  #pragma unroll 8
  for (int j = 0; j < 32; ++j) se += __expf(lg[rrow][colBase + j] - mx);
  se += __shfl_xor(se, 1, 32);
  se += __shfl_xor(se, 2, 32);
  if ((tid & 3) == 0) {
    const size_t o = (size_t)(rowBase + rrow) * NCHUNK + blockIdx.y;
    chunkMax[o] = mx;
    chunkSum[o] = se;
  }
}

// ---------------------------------------------------------------------------
// Kernel 3: combine 782 chunk statistics per row -> per-row loss.
// ---------------------------------------------------------------------------
__global__ __launch_bounds__(256)
void cosface_rowreduce(const float* __restrict__ chunkMax,
                       const float* __restrict__ chunkSum,
                       const float* __restrict__ tgt,
                       float* __restrict__ rowLoss) {
  __shared__ float sm[256];
  const int row = blockIdx.x, tid = threadIdx.x;
  const float* cm = chunkMax + (size_t)row * NCHUNK;
  const float* cs = chunkSum + (size_t)row * NCHUNK;
  float mx = -INFINITY;
  for (int i = tid; i < NCHUNK; i += 256) mx = fmaxf(mx, cm[i]);
  sm[tid] = mx; __syncthreads();
  for (int s = 128; s > 0; s >>= 1) {
    if (tid < s) sm[tid] = fmaxf(sm[tid], sm[tid + s]);
    __syncthreads();
  }
  const float gmax = sm[0];
  __syncthreads();
  float se = 0.0f;
  for (int i = tid; i < NCHUNK; i += 256) se += cs[i] * __expf(cm[i] - gmax);
  sm[tid] = se; __syncthreads();
  for (int s = 128; s > 0; s >>= 1) {
    if (tid < s) sm[tid] += sm[tid + s];
    __syncthreads();
  }
  if (tid == 0) rowLoss[row] = __logf(sm[0]) + gmax - tgt[row];
}

// ---------------------------------------------------------------------------
// Kernel 4: mean over rows -> scalar output.
// ---------------------------------------------------------------------------
__global__ __launch_bounds__(256)
void cosface_mean(const float* __restrict__ rowLoss, float* __restrict__ out) {
  __shared__ float sm[256];
  const int tid = threadIdx.x;
  float s = 0.0f;
  for (int i = tid; i < Bn; i += 256) s += rowLoss[i];
  sm[tid] = s; __syncthreads();
  for (int st = 128; st > 0; st >>= 1) {
    if (tid < st) sm[tid] += sm[tid + st];
    __syncthreads();
  }
  if (tid == 0) out[0] = sm[0] / (float)Bn;
}

// ---------------------------------------------------------------------------
extern "C" void kernel_launch(void* const* d_in, const int* in_sizes, int n_in,
                              void* d_out, int out_size, void* d_ws,
                              size_t ws_size, hipStream_t stream) {
  const float* input  = (const float*)d_in[0];
  const int*   label  = (const int*)d_in[1];
  const float* weight = (const float*)d_in[2];

  // Workspace layout (bytes, all 256B-aligned):
  //   pW       : Cn*Dn*2          = 102,400,000   (bf16 weight, L2-resident)
  //   pA       : Bn*Dn*2          =   1,048,576
  //   chunkMax : Bn*NCHUNK*4      =   3,203,072
  //   chunkSum : Bn*NCHUNK*4      =   3,203,072
  //   tgt      : Bn*4             =       4,096
  //   rowLoss  : Bn*4             =       4,096
  char* ws = (char*)d_ws;
  unsigned short* pW = (unsigned short*)ws;
  unsigned short* pA = (unsigned short*)(ws + (size_t)Cn * Dn * 2);
  float* chunkMax = (float*)(ws + (size_t)Cn * Dn * 2 + (size_t)Bn * Dn * 2);
  float* chunkSum = chunkMax + (size_t)Bn * NCHUNK;
  float* tgt      = chunkSum + (size_t)Bn * NCHUNK;
  float* rowLoss  = tgt + Bn;

  // 1) Normalize + pack both operands to WMMA-ordered bf16.
  pack_l2norm_bf16<<<Cn / 16, 256, 0, stream>>>(weight, pW, /*aLayout=*/0);
  pack_l2norm_bf16<<<Bn / 16, 256, 0, stream>>>(input,  pA, /*aLayout=*/1);

  // 2) Fused WMMA GEMM + margin + per-chunk softmax stats.
  dim3 grid(Bn / ROWS_PB, NCHUNK);
  cosface_gemm<<<grid, 256, 0, stream>>>(pA, pW, label, chunkMax, chunkSum, tgt);

  // 3) Per-row logsumexp combine, 4) mean.
  cosface_rowreduce<<<Bn, 256, 0, stream>>>(chunkMax, chunkSum, tgt, rowLoss);
  cosface_mean<<<1, 256, 0, stream>>>(rowLoss, (float*)d_out);
}
